// Attn_50233937494279
// MI455X (gfx1250) — compile-verified, matrix operand order checked by reading, not run on hardware
//
#include <hip/hip_runtime.h>
#include <cstdint>

// ---------------------------------------------------------------------------
// MI455X (gfx1250) attention block, split-bf16 WMMA GEMMs.
//   G1: X   = q_b @ W_in^T                 [1024,1024] per batch
//   G2: S   = X @ C_b^T   -> d_out(score)  [1024,2048] per batch
//   SM: softmax rows of S in place
//   G3: ctx = P @ C_b                      [1024,1024] per batch
//   G4: out = tanh([ctx|X] @ W_out^T) -> d_out, strided [q,b,d]
// Precision: fp32 operands split into bf16 hi+lo; 3x v_wmma_f32_16x16x32_bf16
// per K=32 chunk reproduces ~fp32 dot-product accuracy.
// Double-buffered LDS (80KB static): one barrier per K-step, LDS stores of
// stage k+1 overlap the WMMAs of stage k.
// ---------------------------------------------------------------------------

typedef __bf16 bf16_t;
typedef __bf16 v16bf __attribute__((ext_vector_type(16)));
typedef __bf16 v8bf  __attribute__((ext_vector_type(8)));
typedef __bf16 v4bf  __attribute__((ext_vector_type(4)));
typedef float  v8f   __attribute__((ext_vector_type(8)));

#define BM 128
#define BN 128
#define BK 32
#define LDSK 40   // padded LDS row pitch in bf16 (80B: 16B-aligned chunks, spread banks)

// Load one 16x32 bf16 WMMA operand fragment from LDS (row-major [*][LDSK]).
// Per CDNA5 ISA layout for 16-bit A/B operands:
//   lanes 0-15 : K = 0..7   in regs 0-3,  K = 16..23 in regs 4-7
//   lanes 16-31: K = 8..15  in regs 0-3,  K = 24..31 in regs 4-7
__device__ __forceinline__ v16bf ld_frag(const bf16_t* base, int row, int kh) {
    const bf16_t* p = base + row * LDSK;
    v8bf c0 = *(const v8bf*)(p + kh);        // ds_load_b128
    v8bf c1 = *(const v8bf*)(p + 16 + kh);   // ds_load_b128
    v16bf r;
#pragma unroll
    for (int i = 0; i < 8; ++i) { r[i] = c0[i]; r[8 + i] = c1[i]; }
    return r;
}

// D[m][n] = sum_k A[m][k] * B'[k][n]  (+ optional tanh epilogue)
//   BT=1 : B stored as [n][k] (dot rows; stride sB)
//   BT=0 : B stored as [k][n] (transposed into LDS on load)
//   SPLIT=1: A source switches from A1 to A2 at k == kSplit (concat without copy)
template <int BT, int TANH, int SPLIT>
__global__ __launch_bounds__(256)
void gemm_bf16x3_kernel(const float* __restrict__ A1, const float* __restrict__ A2,
                        int64_t sA, int64_t bA,
                        const float* __restrict__ B, int64_t sB, int64_t bB,
                        float* __restrict__ D, int64_t sD, int64_t bD,
                        int K, int kSplit)
{
    __shared__ bf16_t sAhi[2][BM * LDSK];
    __shared__ bf16_t sAlo[2][BM * LDSK];
    __shared__ bf16_t sBhi[2][BN * LDSK];
    __shared__ bf16_t sBlo[2][BN * LDSK];

    const int m0 = blockIdx.y * BM;
    const int n0 = blockIdx.x * BN;
    {
        const int64_t b = blockIdx.z;
        A1 += b * bA; A2 += b * bA; B += b * bB; D += b * bD;
    }

    const int t     = threadIdx.x;
    const int lane  = t & 31;
    const int wave  = t >> 5;
    const int waveM = (wave >> 2) * 64;   // 2 waves over M
    const int waveN = (wave & 3) * 32;    // 4 waves over N
    const int lrow  = lane & 15;
    const int kh    = (lane >> 4) * 8;

    v8f acc[4][2] = {};

    const int nk = K / BK;
    float4 av[4], bv[4];

    auto loadA = [&](int kb, float4* v) {
        int k0 = kb * BK;
        const float* Asrc = A1; int kk = k0;
        if (SPLIT && k0 >= kSplit) { Asrc = A2; kk = k0 - kSplit; }
#pragma unroll
        for (int i = 0; i < 4; ++i) {
            int idx = t + i * 256;
            int row = idx >> 3;
            int col = (idx & 7) * 4;
            v[i] = *(const float4*)(Asrc + (int64_t)(m0 + row) * sA + kk + col);
        }
    };
    auto loadB = [&](int kb, float4* v) {
        int k0 = kb * BK;
        if (BT) {
#pragma unroll
            for (int i = 0; i < 4; ++i) {
                int idx = t + i * 256;
                int row = idx >> 3;
                int col = (idx & 7) * 4;
                v[i] = *(const float4*)(B + (int64_t)(n0 + row) * sB + k0 + col);
            }
        } else {
#pragma unroll
            for (int i = 0; i < 4; ++i) {
                int idx = t + i * 256;
                int kr  = idx >> 5;
                int col = (idx & 31) * 4;
                v[i] = *(const float4*)(B + (int64_t)(k0 + kr) * sB + n0 + col);
            }
        }
    };
    auto cvt4 = [](float4 x, v4bf& h, v4bf& l) {
        float xs[4] = {x.x, x.y, x.z, x.w};
#pragma unroll
        for (int j = 0; j < 4; ++j) {
            bf16_t hh = (bf16_t)xs[j];
            h[j] = hh;
            l[j] = (bf16_t)(xs[j] - (float)hh);   // residual -> lo limb
        }
    };
    auto storeA = [&](int buf, const float4* v) {
#pragma unroll
        for (int i = 0; i < 4; ++i) {
            int idx = t + i * 256;
            int row = idx >> 3;
            int col = (idx & 7) * 4;
            v4bf h, l; cvt4(v[i], h, l);
            *(v4bf*)&sAhi[buf][row * LDSK + col] = h;
            *(v4bf*)&sAlo[buf][row * LDSK + col] = l;
        }
    };
    auto storeB = [&](int buf, const float4* v) {
        if (BT) {
#pragma unroll
            for (int i = 0; i < 4; ++i) {
                int idx = t + i * 256;
                int row = idx >> 3;
                int col = (idx & 7) * 4;
                v4bf h, l; cvt4(v[i], h, l);
                *(v4bf*)&sBhi[buf][row * LDSK + col] = h;
                *(v4bf*)&sBlo[buf][row * LDSK + col] = l;
            }
        } else {  // transpose [k][n] tile into [n][k] LDS layout
#pragma unroll
            for (int i = 0; i < 4; ++i) {
                int idx = t + i * 256;
                int kr  = idx >> 5;
                int col = (idx & 31) * 4;
                v4bf h, l; cvt4(v[i], h, l);
#pragma unroll
                for (int j = 0; j < 4; ++j) {
                    sBhi[buf][(col + j) * LDSK + kr] = h[j];
                    sBlo[buf][(col + j) * LDSK + kr] = l[j];
                }
            }
        }
    };
    auto compute = [&](int buf) {
        v16bf bh[2], bl[2];
#pragma unroll
        for (int nt = 0; nt < 2; ++nt) {
            int ncol = waveN + nt * 16 + lrow;
            bh[nt] = ld_frag(sBhi[buf], ncol, kh);
            bl[nt] = ld_frag(sBlo[buf], ncol, kh);
        }
#pragma unroll
        for (int mt = 0; mt < 4; ++mt) {
            int mrow = waveM + mt * 16 + lrow;
            v16bf ah = ld_frag(sAhi[buf], mrow, kh);
            v16bf al = ld_frag(sAlo[buf], mrow, kh);
#pragma unroll
            for (int nt = 0; nt < 2; ++nt) {
                acc[mt][nt] = __builtin_amdgcn_wmma_f32_16x16x32_bf16(
                    false, ah, false, bh[nt], (short)0, acc[mt][nt], false, false);
                acc[mt][nt] = __builtin_amdgcn_wmma_f32_16x16x32_bf16(
                    false, ah, false, bl[nt], (short)0, acc[mt][nt], false, false);
                acc[mt][nt] = __builtin_amdgcn_wmma_f32_16x16x32_bf16(
                    false, al, false, bh[nt], (short)0, acc[mt][nt], false, false);
            }
        }
    };

    // Pipeline: buf(k) is computed while buf(k+1) is being filled; the global
    // loads for k+1 issue before the WMMAs of k so their latency is hidden,
    // and only one barrier per K-step is required.
    loadA(0, av); loadB(0, bv);
    storeA(0, av); storeB(0, bv);
    __syncthreads();
    for (int kb = 0; kb < nk; ++kb) {
        const int cur = kb & 1;
        const bool pref = (kb + 1 < nk);
        if (pref) { loadA(kb + 1, av); loadB(kb + 1, bv); }
        compute(cur);
        if (pref) { storeA(cur ^ 1, av); storeB(cur ^ 1, bv); }
        __syncthreads();
    }

    // Epilogue. C/D layout: reg r -> M = r + 8*(lane>=16), N = lane&15.
    const int rsub = (lane >> 4) * 8;
#pragma unroll
    for (int mt = 0; mt < 4; ++mt) {
#pragma unroll
        for (int nt = 0; nt < 2; ++nt) {
            int row = m0 + waveM + mt * 16 + rsub;
            int col = n0 + waveN + nt * 16 + lrow;
#pragma unroll
            for (int r = 0; r < 8; ++r) {
                float vv = acc[mt][nt][r];
                if (TANH) vv = tanhf(vv);
                D[(int64_t)(row + r) * sD + col] = vv;
            }
        }
    }
}

// In-place softmax over rows of 2048 floats; one 256-thread block per row.
__global__ __launch_bounds__(256)
void softmax2048_kernel(float* __restrict__ S)
{
    float* row = S + (int64_t)blockIdx.x * 2048;
    const int t = threadIdx.x;
    const int lane = t & 31, wave = t >> 5;

    float4 a = *(const float4*)(row + t * 8);
    float4 b = *(const float4*)(row + t * 8 + 4);
    float x[8] = {a.x, a.y, a.z, a.w, b.x, b.y, b.z, b.w};

    float mx = x[0];
#pragma unroll
    for (int i = 1; i < 8; ++i) mx = fmaxf(mx, x[i]);
#pragma unroll
    for (int off = 16; off >= 1; off >>= 1) mx = fmaxf(mx, __shfl_xor(mx, off, 32));

    __shared__ float red[8];
    if (lane == 0) red[wave] = mx;
    __syncthreads();
    float M = red[0];
#pragma unroll
    for (int i = 1; i < 8; ++i) M = fmaxf(M, red[i]);

    float s = 0.f;
#pragma unroll
    for (int i = 0; i < 8; ++i) { x[i] = __expf(x[i] - M); s += x[i]; }
#pragma unroll
    for (int off = 16; off >= 1; off >>= 1) s += __shfl_xor(s, off, 32);
    __syncthreads();
    if (lane == 0) red[wave] = s;
    __syncthreads();
    float tot = 0.f;
#pragma unroll
    for (int i = 0; i < 8; ++i) tot += red[i];
    float inv = 1.0f / tot;

    *(float4*)(row + t * 8)     = make_float4(x[0]*inv, x[1]*inv, x[2]*inv, x[3]*inv);
    *(float4*)(row + t * 8 + 4) = make_float4(x[4]*inv, x[5]*inv, x[6]*inv, x[7]*inv);
}

extern "C" void kernel_launch(void* const* d_in, const int* in_sizes, int n_in,
                              void* d_out, int out_size, void* d_ws, size_t ws_size,
                              hipStream_t stream) {
    (void)in_sizes; (void)n_in; (void)out_size; (void)ws_size;

    const float* q     = (const float*)d_in[0];   // [1024, 16, 1024]
    const float* c     = (const float*)d_in[1];   // [2048, 16, 1024]
    const float* W_in  = (const float*)d_in[2];   // [1024, 1024]
    const float* W_out = (const float*)d_in[3];   // [1024, 2048]

    const int B = 16, Q = 1024, CL = 2048, DIM = 1024;

    float* out   = (float*)d_out;                          // [Q, B, DIM]
    float* score = out + (int64_t)Q * B * DIM;             // [B, Q, CL], softmaxed in place
    float* X     = (float*)d_ws;                           // [B, Q, DIM]  (64 MB)
    float* ctx   = X + (int64_t)B * Q * DIM;               // [B, Q, DIM]  (64 MB)

    dim3 blk(256);

    // G1: X_b = q_b @ W_in^T   (A rows strided through [q,b,d] layout)
    gemm_bf16x3_kernel<1, 0, 0><<<dim3(DIM / BN, Q / BM, B), blk, 0, stream>>>(
        q, q, (int64_t)B * DIM, (int64_t)DIM,
        W_in, (int64_t)DIM, 0,
        X, (int64_t)DIM, (int64_t)Q * DIM,
        DIM, 0);

    // G2: logits S_b = X_b @ C_b^T -> score region of d_out
    gemm_bf16x3_kernel<1, 0, 0><<<dim3(CL / BN, Q / BM, B), blk, 0, stream>>>(
        X, X, (int64_t)DIM, (int64_t)Q * DIM,
        c, (int64_t)B * DIM, (int64_t)DIM,
        score, (int64_t)CL, (int64_t)Q * CL,
        DIM, 0);

    // Row softmax in place (score is a required output)
    softmax2048_kernel<<<B * Q, blk, 0, stream>>>(score);

    // G3: ctx_b = P_b @ C_b   (B is [k][n] through the [c,b,d] layout)
    gemm_bf16x3_kernel<0, 0, 0><<<dim3(DIM / BN, Q / BM, B), blk, 0, stream>>>(
        score, score, (int64_t)CL, (int64_t)Q * CL,
        c, (int64_t)B * DIM, (int64_t)DIM,
        ctx, (int64_t)DIM, (int64_t)Q * DIM,
        CL, 0);

    // G4: out = tanh([ctx | X] @ W_out^T), written strided as [q, b, d]
    gemm_bf16x3_kernel<1, 1, 1><<<dim3(DIM / BN, Q / BM, B), blk, 0, stream>>>(
        ctx, X, (int64_t)DIM, (int64_t)Q * DIM,
        W_out, (int64_t)(2 * DIM), 0,
        out, (int64_t)B * DIM, (int64_t)DIM,
        2 * DIM, DIM);
}
